// ReadthroughModelV2_23776938950901
// MI455X (gfx1250) — matrix operand order repaired and checked by reading, not compiled, not run on hardware
//
#include <hip/hip_runtime.h>

// ---------------- model dims ----------------
#define BB 16
#define SS 2048
#define MROWS (BB*SS)   // 32768
#define DD 512
#define HH 8
#define HDD 64

typedef __attribute__((ext_vector_type(16))) __bf16 v16bf;
typedef __attribute__((ext_vector_type(8)))  float  v8f;

#if defined(__has_builtin)
#if __has_builtin(__builtin_amdgcn_tensor_load_to_lds)
#define HAVE_TDM 1
#endif
#endif
#ifndef HAVE_TDM
#define HAVE_TDM 0
#endif

#if HAVE_TDM
typedef __attribute__((ext_vector_type(4))) unsigned int v4u;
typedef __attribute__((ext_vector_type(4))) int v4i;
typedef __attribute__((ext_vector_type(8))) int v8i;
#endif

union AFrag {
    v16bf v;
    unsigned short u[16];
    uint4 q[2];
};

static __device__ __forceinline__ unsigned short f2bf(float f) {
    unsigned int u = __float_as_uint(f);
    unsigned int r = u + 0x7FFFu + ((u >> 16) & 1u);   // round-to-nearest-even
    return (unsigned short)(r >> 16);
}
static __device__ __forceinline__ float bf2f(unsigned short h) {
    return __uint_as_float(((unsigned int)h) << 16);
}
static __device__ __forceinline__ float gelu_exact(float x) {
    return 0.5f * x * (1.0f + erff(x * 0.7071067811865475f));
}
static __device__ __forceinline__ v8f vzero8() {
    v8f z = {0.f,0.f,0.f,0.f,0.f,0.f,0.f,0.f};
    return z;
}

// ---------------- f32 -> bf16 convert ----------------
__global__ void f2bf_kernel(const float* __restrict__ in, unsigned short* __restrict__ out, int n) {
    int i = blockIdx.x * 256 + threadIdx.x;
    if (i < n) out[i] = f2bf(in[i]);
}

// ---------------- f32 [K,N] -> bf16 [N,K] transpose-convert (conv weights) ----------------
__global__ void f2bf_transpose_kernel(const float* __restrict__ in, unsigned short* __restrict__ out,
                                      int K, int N) {
    int i = blockIdx.x * 256 + threadIdx.x;
    if (i < K * N) {
        int k = i / N, n = i - k * N;
        out[(size_t)n * K + k] = f2bf(in[i]);
    }
}

// ---------------- embedding gather ----------------
__global__ void embed_kernel(const int* __restrict__ tok, const int* __restrict__ pos,
                             const float* __restrict__ nt, const float* __restrict__ pe,
                             unsigned short* __restrict__ out) {
    int t = blockIdx.x;        // 0..MROWS-1
    int c = threadIdx.x;       // 0..127
    int tk = tok[t];
    int pi = pos[t] + 1022;    // UP = 1022
    if (pi < 0) pi = 0;
    float v = nt[tk * 128 + c] + pe[(size_t)pi * 128 + c];
    out[(size_t)t * 128 + c] = f2bf(v);
}

// ---------------- generic WMMA GEMM: Y = act(A @ W^T + bias) (+res) ----------------
// A: bf16 [M,K] row-major; taps>1 does the im2col gather over length-SS sequences.
// W: bf16 [N,K] row-major always (conv weights pre-transposed on host side).
__global__ __launch_bounds__(128)
void gemm_kernel(const unsigned short* __restrict__ A,
                 const unsigned short* __restrict__ W,
                 const float* __restrict__ bias,
                 const float* __restrict__ residual,
                 float* __restrict__ outF,
                 unsigned short* __restrict__ outB,
                 int M, int N, int K,
                 int cin, int taps, int center, int act) {
    const int lane = threadIdx.x & 31;
    const int wave = threadIdx.x >> 5;
    const int hi   = lane >> 4;          // half-wave select
    const int ln   = lane & 15;
    const int m0   = blockIdx.x * 64 + wave * 16;
    const int n0   = blockIdx.y * 64;

    const int m = m0 + ln;               // A-row carried by this lane
    const int s = m % SS;
    const int rowbase = m - s;           // b*SS

    v8f acc[4] = {vzero8(), vzero8(), vzero8(), vzero8()};

    for (int k0 = 0; k0 < K; k0 += 32) {
        // ---- A fragment (16x32 bf16, CDNA5 swizzle) ----
        AFrag a;
#pragma unroll
        for (int g = 0; g < 2; ++g) {
            int ks = k0 + (hi ? 8 : 0) + g * 16;   // 8 contiguous k per group
            int tap, c;
            if (taps > 1) { tap = ks / cin; c = ks - tap * cin; }
            else          { tap = 0;       c = ks; }
            int sp = s + tap - center;
            if ((unsigned)sp < (unsigned)SS) {
                a.q[g] = *reinterpret_cast<const uint4*>(A + (size_t)(rowbase + sp) * cin + c);
            } else {
                a.q[g] = make_uint4(0u, 0u, 0u, 0u);
            }
        }
        {
            int kp = (k0 + 32 < K) ? (k0 + 32) : k0;
            __builtin_prefetch(W + (size_t)(n0 + ln) * K + kp, 0, 0);
        }
        // ---- B fragments (32x16 each) + WMMA ----
#pragma unroll
        for (int t = 0; t < 4; ++t) {
            AFrag b;
            int n  = n0 + t * 16 + ln;
            int kk = k0 + (hi ? 8 : 0);
            const uint4* p = reinterpret_cast<const uint4*>(W + (size_t)n * K + kk);
            b.q[0] = p[0];
            b.q[1] = p[2];
            acc[t] = __builtin_amdgcn_wmma_f32_16x16x32_bf16(
                false, a.v, false, b.v, (short)0, acc[t], false, false);
        }
    }

    // ---- epilogue: bias, act, residual, stores ----
#pragma unroll
    for (int t = 0; t < 4; ++t) {
        int n = n0 + t * 16 + ln;
        float bv = bias ? bias[n] : 0.0f;
#pragma unroll
        for (int j = 0; j < 8; ++j) {
            int row = m0 + j + hi * 8;
            float v = acc[t][j] + bv;
            if (act) v = gelu_exact(v);
            size_t idx = (size_t)row * N + n;
            if (residual) v += residual[idx];
            if (outF) outF[idx] = v;
            if (outB) outB[idx] = f2bf(v);
        }
    }
}

// ---------------- fused (gelu ->) LayerNorm over channels ----------------
__global__ __launch_bounds__(256)
void gelu_ln_kernel(const float* __restrict__ in, const float* __restrict__ g,
                    const float* __restrict__ b, float* __restrict__ outF,
                    unsigned short* __restrict__ outB, int C, int do_gelu) {
    __shared__ float red[256];
    const int row = blockIdx.x;
    const int tid = threadIdx.x;
    const float* x = in + (size_t)row * C;
    const int nch = C >> 8;              // 1 (C=256) or 2 (C=512)
    float loc[2];
    float s = 0.f;
    for (int i = 0; i < nch; ++i) {
        float v = x[tid + i * 256];
        if (do_gelu) v = gelu_exact(v);
        loc[i] = v;
        s += v;
    }
    red[tid] = s;
    __syncthreads();
    for (int o = 128; o > 0; o >>= 1) {
        if (tid < o) red[tid] += red[tid + o];
        __syncthreads();
    }
    float mean = red[0] / (float)C;
    __syncthreads();
    float vs = 0.f;
    for (int i = 0; i < nch; ++i) { float d = loc[i] - mean; vs += d * d; }
    red[tid] = vs;
    __syncthreads();
    for (int o = 128; o > 0; o >>= 1) {
        if (tid < o) red[tid] += red[tid + o];
        __syncthreads();
    }
    float inv = rsqrtf(red[0] / (float)C + 1e-5f);
    for (int i = 0; i < nch; ++i) {
        int c = tid + i * 256;
        float v = (loc[i] - mean) * inv * g[c] + b[c];
        if (outF) outF[(size_t)row * C + c] = v;
        if (outB) outB[(size_t)row * C + c] = f2bf(v);
    }
}

// ---------------- banded local attention, fully fused, WMMA both GEMMs ----------------
// grid: (SS/64, H, B), block: 128 (4 waves). Each wave owns 16 query rows.
#define KV_STRIDE 72
#define VT_STRIDE 208
#define AT_STRIDE 208

__global__ __launch_bounds__(128)
void attn_kernel(const unsigned short* __restrict__ qkv, unsigned short* __restrict__ out) {
    __shared__ unsigned short kv_lds[192 * KV_STRIDE];       // K window, later V^T
    __shared__ unsigned short at_lds[4][16 * AT_STRIDE];     // attn probs (bf16) per wave

    const int blk = blockIdx.x;
    const int h   = blockIdx.y;
    const int bb  = blockIdx.z;
    const int tid = threadIdx.x;
    const int lane = tid & 31, wave = tid >> 5;
    const int hi = lane >> 4, ln = lane & 15;
    const int qb0 = blk * 64;
    const size_t baseRow = (size_t)bb * SS;

    // ---- stage K window [192 keys][64 hd] into LDS ----
#if HAVE_TDM
    const bool interior = (blk > 0) && (blk < (SS / 64) - 1);
    if (interior) {
        // 2-D Tensor Data Mover tile: 192 rows x 64 bf16, row stride 1536 elements.
        // TDM padding (32 DWORDs data + 4 DWORDs pad) reproduces the 72-element LDS stride.
        if (wave == 0) {
            unsigned long long gaddr =
                (unsigned long long)(const void*)(qkv + (baseRow + qb0 - 64) * 1536 + 512 + h * 64);
            unsigned int laddr = (unsigned int)(unsigned long long)(void*)&kv_lds[0];
            v4u g0;
            g0.x = 1u;                                           // count = 1 (valid)
            g0.y = laddr;                                        // lds_addr
            g0.z = (unsigned int)(gaddr & 0xFFFFFFFFu);          // global_addr[31:0]
            g0.w = (unsigned int)((gaddr >> 32) & 0x01FFFFFFu)   // global_addr[56:32]
                   | (2u << 30);                                 // type = 2 (image)
            v8i g1;
            g1[0] = (1 << 16)       // data_size = 2 bytes
                  | (1 << 20)       // pad_enable
                  | (4 << 22)       // pad_interval: 32 DWORDs
                  | (3 << 25);      // pad_amount: 4 DWORDs
            g1[1] = 0;                               // dim0[15:0]=0 (dim0 = 2^30)
            g1[2] = 0x4000;                          // dim0[31:16]; dim1[15:0]=0 (dim1 = 2^30)
            g1[3] = 0x4000 | (64 << 16);             // dim1[31:16]; tile_dim0 = 64
            g1[4] = 192;                             // tile_dim1 = 192, tile_dim2 = 0
            g1[5] = 1536;                            // tensor_dim0_stride[31:0]
            g1[6] = 0;                               // stride0[47:32], stride1[15:0]
            g1[7] = 0;                               // stride1[47:16]
            v4i gz = {0, 0, 0, 0};
#if __clang_major__ >= 23
            v8i gz8 = {0, 0, 0, 0, 0, 0, 0, 0};
            __builtin_amdgcn_tensor_load_to_lds(g0, g1, gz, gz, gz8, 0);
#else
            __builtin_amdgcn_tensor_load_to_lds(g0, g1, gz, gz, 0);
#endif
#if __has_builtin(__builtin_amdgcn_s_wait_tensorcnt)
            __builtin_amdgcn_s_wait_tensorcnt(0);
#else
            asm volatile("s_wait_tensorcnt 0x0" ::: "memory");
#endif
        }
    } else
#endif
    {
        for (int i = tid; i < 192 * 64; i += 128) {
            int r = i >> 6, c = i & 63;
            int kg = qb0 - 64 + r;
            unsigned short v = 0;
            if ((unsigned)kg < (unsigned)SS)
                v = qkv[(baseRow + kg) * 1536 + 512 + h * 64 + c];
            kv_lds[r * KV_STRIDE + c] = v;
        }
    }
    __syncthreads();

    // ---- Q fragments from global (K = 64 -> 2 steps) ----
    AFrag af[2];
    {
        int qrow = qb0 + wave * 16 + ln;
        const unsigned short* qp = qkv + (baseRow + qrow) * 1536 + h * 64;
#pragma unroll
        for (int kstep = 0; kstep < 2; ++kstep) {
#pragma unroll
            for (int g = 0; g < 2; ++g) {
                int ks = kstep * 32 + (hi ? 8 : 0) + g * 16;
                af[kstep].q[g] = *reinterpret_cast<const uint4*>(qp + ks);
            }
        }
    }

    // ---- scores: 16 x 192 per wave (12 N-tiles) ----
    v8f sacc[12];
#pragma unroll
    for (int t = 0; t < 12; ++t) sacc[t] = vzero8();
#pragma unroll
    for (int t = 0; t < 12; ++t) {
#pragma unroll
        for (int kstep = 0; kstep < 2; ++kstep) {
            AFrag bf_;
            int keyr = t * 16 + ln;
            int kk = kstep * 32 + (hi ? 8 : 0);
            const uint4* p = reinterpret_cast<const uint4*>(&kv_lds[keyr * KV_STRIDE + kk]);
            bf_.q[0] = p[0];
            bf_.q[1] = p[2];
            sacc[t] = __builtin_amdgcn_wmma_f32_16x16x32_bf16(
                false, af[kstep].v, false, bf_.v, (short)0, sacc[t], false, false);
        }
    }

    // ---- masked softmax (band +/-64, seq bounds), half-wave shuffle reductions ----
    const float scale = 0.125f;  // 1/sqrt(64)
    unsigned short* arow = at_lds[wave];
#pragma unroll
    for (int j = 0; j < 8; ++j) {
        int rl = wave * 16 + j + hi * 8;       // local query row 0..63
        float mx = -3.0e38f;
#pragma unroll
        for (int t = 0; t < 12; ++t) {
            int c = t * 16 + ln;
            int kg = qb0 - 64 + c;
            bool ok = (c >= rl) && (c <= rl + 128) && ((unsigned)kg < (unsigned)SS);
            float v = ok ? sacc[t][j] * scale : -3.0e38f;
            sacc[t][j] = v;
            mx = fmaxf(mx, v);
        }
#pragma unroll
        for (int o = 8; o > 0; o >>= 1) mx = fmaxf(mx, __shfl_xor(mx, o, 16));
        float sum = 0.f;
#pragma unroll
        for (int t = 0; t < 12; ++t) {
            float p = __expf(sacc[t][j] - mx);
            sacc[t][j] = p;
            sum += p;
        }
#pragma unroll
        for (int o = 8; o > 0; o >>= 1) sum += __shfl_xor(sum, o, 16);
        float rinv = 1.0f / sum;
        int lrow = j + hi * 8;
#pragma unroll
        for (int t = 0; t < 12; ++t)
            arow[lrow * AT_STRIDE + t * 16 + ln] = f2bf(sacc[t][j] * rinv);
    }
    __syncthreads();

    // ---- restage V transposed [64 hd][192 keys] into same LDS ----
    for (int i = tid; i < 192 * 64; i += 128) {
        int r = i >> 6, c = i & 63;
        int kg = qb0 - 64 + r;
        unsigned short v = 0;
        if ((unsigned)kg < (unsigned)SS)
            v = qkv[(baseRow + kg) * 1536 + 1024 + h * 64 + c];
        kv_lds[c * VT_STRIDE + r] = v;
    }
    __syncthreads();

    // ---- out = P @ V : 16 x 64 per wave, K = 192 (6 steps) ----
    v8f oacc[4] = {vzero8(), vzero8(), vzero8(), vzero8()};
#pragma unroll
    for (int kstep = 0; kstep < 6; ++kstep) {
        int k0 = kstep * 32;
        AFrag a;
#pragma unroll
        for (int g = 0; g < 2; ++g) {
            int ks = k0 + (hi ? 8 : 0) + g * 16;
            a.q[g] = *reinterpret_cast<const uint4*>(&arow[ln * AT_STRIDE + ks]);
        }
#pragma unroll
        for (int t = 0; t < 4; ++t) {
            AFrag bf_;
            int cc = t * 16 + ln;                 // hd column
            int kk = k0 + (hi ? 8 : 0);
            const uint4* p = reinterpret_cast<const uint4*>(&kv_lds[cc * VT_STRIDE + kk]);
            bf_.q[0] = p[0];
            bf_.q[1] = p[2];
            oacc[t] = __builtin_amdgcn_wmma_f32_16x16x32_bf16(
                false, a.v, false, bf_.v, (short)0, oacc[t], false, false);
        }
    }
#pragma unroll
    for (int t = 0; t < 4; ++t) {
        int col = h * 64 + t * 16 + ln;
#pragma unroll
        for (int j = 0; j < 8; ++j) {
            int row = qb0 + wave * 16 + j + hi * 8;
            out[(baseRow + row) * 512 + col] = f2bf(oacc[t][j]);
        }
    }
}

// ---------------- pooling: q projection of learned query ----------------
__global__ void pool_q_kernel(const float* __restrict__ pq, const float* __restrict__ piw,
                              const float* __restrict__ pib, float* __restrict__ qout) {
    int o = blockIdx.x * 64 + threadIdx.x;   // 512 outputs
    float s = pib[o];
    for (int c = 0; c < 512; ++c) s += pq[c] * piw[(size_t)o * 512 + c];
    qout[o] = s;
}

// ---------------- pooling: single-query cross attention over S ----------------
__global__ __launch_bounds__(256)
void pool_attn_kernel(const unsigned short* __restrict__ kv, const float* __restrict__ q,
                      float* __restrict__ pooled) {
    __shared__ float sc[SS];
    __shared__ float red[256];
    const int h = blockIdx.x, b = blockIdx.y;
    const int tid = threadIdx.x;
    const float scale = 0.125f;
    for (int s = tid; s < SS; s += 256) {
        const unsigned short* kp = kv + ((size_t)b * SS + s) * 1024 + h * 64;
        float acc = 0.f;
        for (int c = 0; c < 64; ++c) acc += q[h * 64 + c] * bf2f(kp[c]);
        sc[s] = acc * scale;
    }
    __syncthreads();
    float mx = -3.0e38f;
    for (int s = tid; s < SS; s += 256) mx = fmaxf(mx, sc[s]);
    red[tid] = mx;
    __syncthreads();
    for (int o = 128; o > 0; o >>= 1) {
        if (tid < o) red[tid] = fmaxf(red[tid], red[tid + o]);
        __syncthreads();
    }
    mx = red[0];
    __syncthreads();
    float sm = 0.f;
    for (int s = tid; s < SS; s += 256) { float p = __expf(sc[s] - mx); sc[s] = p; sm += p; }
    red[tid] = sm;
    __syncthreads();
    for (int o = 128; o > 0; o >>= 1) {
        if (tid < o) red[tid] += red[tid + o];
        __syncthreads();
    }
    float rinv = 1.0f / red[0];
    __syncthreads();
    if (tid < 64) {
        int c = tid;
        float acc = 0.f;
        for (int s = 0; s < SS; ++s)
            acc += sc[s] * bf2f(kv[((size_t)b * SS + s) * 1024 + 512 + h * 64 + c]);
        pooled[(size_t)b * 512 + h * 64 + c] = acc * rinv;
    }
}

// ---------------- pooled out-proj + LN + head MLP ----------------
__global__ __launch_bounds__(256)
void pool_tail_kernel(const float* __restrict__ pooled, const float* __restrict__ pow_,
                      const float* __restrict__ pob, const float* __restrict__ pg,
                      const float* __restrict__ pb, const float* __restrict__ hw1,
                      const float* __restrict__ hb1, const float* __restrict__ hw2,
                      const float* __restrict__ hb2, float* __restrict__ outp) {
    __shared__ float xrow[512];
    __shared__ float y[512];
    __shared__ float red[256];
    const int b = blockIdx.x, tid = threadIdx.x;
    for (int c = tid; c < 512; c += 256) xrow[c] = pooled[(size_t)b * 512 + c];
    __syncthreads();
    for (int i = 0; i < 2; ++i) {
        int o = tid + i * 256;
        float s = pob[o];
        for (int c = 0; c < 512; ++c) s += xrow[c] * pow_[(size_t)o * 512 + c];
        y[o] = s;
    }
    __syncthreads();
    float s0 = 0.f;
    for (int i = 0; i < 2; ++i) s0 += y[tid + i * 256];
    red[tid] = s0;
    __syncthreads();
    for (int o = 128; o > 0; o >>= 1) {
        if (tid < o) red[tid] += red[tid + o];
        __syncthreads();
    }
    float mean = red[0] / 512.f;
    __syncthreads();
    float v0 = 0.f;
    for (int i = 0; i < 2; ++i) { float d = y[tid + i * 256] - mean; v0 += d * d; }
    red[tid] = v0;
    __syncthreads();
    for (int o = 128; o > 0; o >>= 1) {
        if (tid < o) red[tid] += red[tid + o];
        __syncthreads();
    }
    float inv = rsqrtf(red[0] / 512.f + 1e-5f);
    __syncthreads();
    for (int i = 0; i < 2; ++i) {
        int c = tid + i * 256;
        y[c] = (y[c] - mean) * inv * pg[c] + pb[c];
    }
    __syncthreads();
    if (tid < 32) {
        float s = hb1[tid];
        for (int c = 0; c < 512; ++c) s += y[c] * hw1[(size_t)tid * 512 + c];
        float hv = gelu_exact(s) * hw2[tid];
        for (int o = 16; o > 0; o >>= 1) hv += __shfl_xor(hv, o, 32);
        if (tid == 0) outp[b] = hv + hb2[0];
    }
}

// ---------------- host orchestration ----------------
extern "C" void kernel_launch(void* const* d_in, const int* in_sizes, int n_in,
                              void* d_out, int out_size, void* d_ws, size_t ws_size,
                              hipStream_t stream) {
    (void)in_sizes; (void)n_in; (void)out_size; (void)ws_size;
    const int*   tokens    = (const int*)d_in[0];
    const int*   positions = (const int*)d_in[1];
    const float* nt_emb = (const float*)d_in[2];
    const float* pos_emb = (const float*)d_in[3];
    const float* c1_w = (const float*)d_in[4];
    const float* c1_g = (const float*)d_in[5];
    const float* c1_b = (const float*)d_in[6];
    const float* c2_w = (const float*)d_in[7];
    const float* c2_g = (const float*)d_in[8];
    const float* c2_b = (const float*)d_in[9];
    const float* t_g1 = (const float*)d_in[10];
    const float* t_b1 = (const float*)d_in[11];
    const float* t_iw = (const float*)d_in[12];
    const float* t_ib = (const float*)d_in[13];
    const float* t_ow = (const float*)d_in[14];
    const float* t_ob = (const float*)d_in[15];
    const float* t_g2 = (const float*)d_in[16];
    const float* t_b2 = (const float*)d_in[17];
    const float* t_w1 = (const float*)d_in[18];
    const float* t_bb1 = (const float*)d_in[19];
    const float* t_w2 = (const float*)d_in[20];
    const float* t_bb2 = (const float*)d_in[21];
    const float* pq   = (const float*)d_in[22];
    const float* p_iw = (const float*)d_in[23];
    const float* p_ib = (const float*)d_in[24];
    const float* p_ow = (const float*)d_in[25];
    const float* p_ob = (const float*)d_in[26];
    const float* p_g  = (const float*)d_in[27];
    const float* p_b  = (const float*)d_in[28];
    const float* h_w1 = (const float*)d_in[29];
    const float* h_b1 = (const float*)d_in[30];
    const float* h_w2 = (const float*)d_in[31];
    const float* h_b2 = (const float*)d_in[32];
    float* outp = (float*)d_out;

    char* ws = (char*)d_ws;
    size_t off = 0;
    auto alloc = [&](size_t bytes) -> char* {
        char* p = ws + off;
        off += (bytes + 255) & ~(size_t)255;
        return p;
    };
    unsigned short* xemb = (unsigned short*)alloc((size_t)MROWS * 128 * 2);
    unsigned short* c1wb = (unsigned short*)alloc((size_t)640 * 256 * 2);   // [256][640] NK
    unsigned short* c2wb = (unsigned short*)alloc((size_t)768 * 512 * 2);   // [512][768] NK
    unsigned short* iwb  = (unsigned short*)alloc((size_t)2 * 1536 * 512 * 2);
    unsigned short* owb  = (unsigned short*)alloc((size_t)2 * 512 * 512 * 2);
    unsigned short* w1b  = (unsigned short*)alloc((size_t)2 * 2048 * 512 * 2);
    unsigned short* w2b  = (unsigned short*)alloc((size_t)2 * 512 * 2048 * 2);
    unsigned short* pkvb = (unsigned short*)alloc((size_t)1024 * 512 * 2);
    float* xf   = (float*)alloc((size_t)MROWS * 512 * 4);
    float* tmpf = (float*)alloc((size_t)MROWS * 512 * 4);
    unsigned short* lnb  = (unsigned short*)alloc((size_t)MROWS * 512 * 2);   // ln out / attn out / x bf16
    unsigned short* qkvb = (unsigned short*)alloc((size_t)MROWS * 1536 * 2);  // qkv / pool kv
    unsigned short* hb   = (unsigned short*)alloc((size_t)MROWS * 2048 * 2);  // ffn hidden
    float* qpool  = (float*)alloc(512 * 4);
    float* pooled = (float*)alloc((size_t)BB * 512 * 4);

    auto conv = [&](const float* src, unsigned short* dst, int n) {
        f2bf_kernel<<<(n + 255) / 256, 256, 0, stream>>>(src, dst, n);
    };
    // conv weights [K,N] -> [N,K] bf16 so every GEMM uses contiguous B loads
    f2bf_transpose_kernel<<<(640 * 256 + 255) / 256, 256, 0, stream>>>(c1_w, c1wb, 640, 256);
    f2bf_transpose_kernel<<<(768 * 512 + 255) / 256, 256, 0, stream>>>(c2_w, c2wb, 768, 512);
    conv(t_iw, iwb, 2 * 1536 * 512);
    conv(t_ow, owb, 2 * 512 * 512);
    conv(t_w1, w1b, 2 * 2048 * 512);
    conv(t_w2, w2b, 2 * 512 * 2048);
    conv(p_iw + 512 * 512, pkvb, 1024 * 512);   // wk;wv rows of p_iw

    embed_kernel<<<MROWS, 128, 0, stream>>>(tokens, positions, nt_emb, pos_emb, xemb);

    // conv1 (k=5, 128->256) as GEMM + gelu-LN
    gemm_kernel<<<dim3(MROWS / 64, 256 / 64), 128, 0, stream>>>(
        xemb, c1wb, nullptr, nullptr, tmpf, nullptr, MROWS, 256, 640, 128, 5, 2, 0);
    gelu_ln_kernel<<<MROWS, 256, 0, stream>>>(tmpf, c1_g, c1_b, nullptr, lnb, 256, 1);

    // conv2 (k=3, 256->512) as GEMM + gelu-LN -> master x (f32)
    gemm_kernel<<<dim3(MROWS / 64, 512 / 64), 128, 0, stream>>>(
        lnb, c2wb, nullptr, nullptr, tmpf, nullptr, MROWS, 512, 768, 256, 3, 1, 0);
    gelu_ln_kernel<<<MROWS, 256, 0, stream>>>(tmpf, c2_g, c2_b, xf, nullptr, 512, 1);

    // 2 encoder layers
    for (int i = 0; i < 2; ++i) {
        gelu_ln_kernel<<<MROWS, 256, 0, stream>>>(xf, t_g1 + i * 512, t_b1 + i * 512,
                                                  nullptr, lnb, 512, 0);
        gemm_kernel<<<dim3(MROWS / 64, 1536 / 64), 128, 0, stream>>>(
            lnb, iwb + (size_t)i * 1536 * 512, t_ib + i * 1536, nullptr, nullptr, qkvb,
            MROWS, 1536, 512, 512, 1, 0, 0);
        attn_kernel<<<dim3(SS / 64, HH, BB), 128, 0, stream>>>(qkvb, lnb);
        gemm_kernel<<<dim3(MROWS / 64, 512 / 64), 128, 0, stream>>>(
            lnb, owb + (size_t)i * 512 * 512, t_ob + i * 512, xf, xf, nullptr,
            MROWS, 512, 512, 512, 1, 0, 0);
        gelu_ln_kernel<<<MROWS, 256, 0, stream>>>(xf, t_g2 + i * 512, t_b2 + i * 512,
                                                  nullptr, lnb, 512, 0);
        gemm_kernel<<<dim3(MROWS / 64, 2048 / 64), 128, 0, stream>>>(
            lnb, w1b + (size_t)i * 2048 * 512, t_bb1 + i * 2048, nullptr, nullptr, hb,
            MROWS, 2048, 512, 512, 1, 0, 1);
        gemm_kernel<<<dim3(MROWS / 64, 512 / 64), 128, 0, stream>>>(
            hb, w2b + (size_t)i * 512 * 2048, t_bb2 + i * 512, xf, xf, nullptr,
            MROWS, 512, 2048, 2048, 1, 0, 0);
    }

    // attention pooling
    f2bf_kernel<<<(MROWS * 512 + 255) / 256, 256, 0, stream>>>(xf, lnb, MROWS * 512);
    gemm_kernel<<<dim3(MROWS / 64, 1024 / 64), 128, 0, stream>>>(
        lnb, pkvb, p_ib + 512, nullptr, nullptr, qkvb, MROWS, 1024, 512, 512, 1, 0, 0);
    pool_q_kernel<<<8, 64, 0, stream>>>(pq, p_iw, p_ib, qpool);
    pool_attn_kernel<<<dim3(HH, BB), 256, 0, stream>>>(qkvb, qpool, pooled);
    pool_tail_kernel<<<BB, 256, 0, stream>>>(pooled, p_ow, p_ob, p_g, p_b,
                                             h_w1, h_b1, h_w2, h_b2, outp);
}
